// KIM_76501957476762
// MI455X (gfx1250) — matrix operand order repaired and checked by reading, not compiled, not run on hardware
//
#include <hip/hip_runtime.h>
#include <math.h>

// ---------------------------------------------------------------------------
// KIM news-recommender forward for MI455X (gfx1250, wave32, WMMA + TDM).
// Dense GEMMs: Tensor Data Mover stages f32 tiles into LDS (double-buffered,
// OOB zero-fill in HW, TENSORcnt-tracked), lanes convert to bf16 in WMMA
// fragment layout, compute on v_wmma_f32_16x16x32_bf16 (f32 accumulate).
// Workspace requirement: ~134 MB (persistent ~45.8 MB + arena ~88.3 MB).
// ---------------------------------------------------------------------------

typedef __attribute__((ext_vector_type(16))) __bf16 v16bf;
typedef __attribute__((ext_vector_type(8)))  float  v8f;
typedef unsigned int u32x4 __attribute__((ext_vector_type(4)));
typedef int          i32x4 __attribute__((ext_vector_type(4)));
typedef int          i32x8 __attribute__((ext_vector_type(8)));

#if __has_builtin(__builtin_amdgcn_tensor_load_to_lds) && \
    __has_builtin(__builtin_amdgcn_s_wait_tensorcnt)
#define KIM_HAS_TDM 1
#else
#define KIM_HAS_TDM 0
#endif

static constexpr int BSZ   = 16;   // batch
static constexpr int SNUM  = 5;    // candidate samples
static constexpr int MC    = 50;   // max clicked
static constexpr int WN    = 23;   // words per title
static constexpr int WD    = 300;  // word dim
static constexpr int AD    = 400;  // attention dim
static constexpr int HEADS = 20;
static constexpr int HD    = 20;   // head dim
static constexpr int EN    = 5;    // entities
static constexpr int NN    = 5;    // neighbors
static constexpr int ED    = 100;  // entity dim

static __device__ __forceinline__ unsigned short kim_f2bf(float f) {
  unsigned int u = __float_as_uint(f);
  u += 0x7FFFu + ((u >> 16) & 1u);           // round-to-nearest-even
  return (unsigned short)(u >> 16);
}

// low 32 bits of a generic LDS address == raw LDS byte offset (aperture is in
// the high half per CDNA5 addressing rules)
static __device__ __forceinline__ unsigned kim_lds_off(const void* p) {
  return (unsigned)(unsigned long long)p;
}

#if KIM_HAS_TDM
// Issue a 2D TDM load: tile_x contiguous f32 elements * tile_y rows, row
// stride row_stride_elems, into LDS byte offset lds_byte. tensor_x/tensor_y
// bound the in-range extents (HW returns zeros beyond them).
static __device__ __forceinline__ void kim_tdm_load2d(
    const float* gptr, unsigned lds_byte, int tile_x, int tile_y,
    long long tensor_x, long long tensor_y, long long row_stride_elems)
{
  unsigned long long ga = (unsigned long long)gptr;
  u32x4 g0;
  g0.x = 1u;                                          // count=1 valid descriptor
  g0.y = lds_byte;                                    // lds_addr [63:32]
  g0.z = (unsigned)(ga & 0xFFFFFFFFull);              // global_addr lo
  g0.w = (unsigned)((ga >> 32) & 0x01FFFFFFull) | (2u << 30);  // addr hi | type=2
  unsigned long long td0 = (unsigned long long)(tensor_x < 0 ? 0 : tensor_x);
  unsigned long long td1 = (unsigned long long)(tensor_y < 0 ? 0 : tensor_y);
  unsigned long long st  = (unsigned long long)row_stride_elems;
  union { unsigned long long q[4]; i32x8 v; } g1;
  // [17:16]=data_size(4B); [79:48]=tensor_dim0; [111:80]=tensor_dim1;
  // [127:112]=tile_dim0; [143:128]=tile_dim1; [207:160]=tensor_dim0_stride
  g1.q[0] = (2ull << 16) | ((td0 & 0xFFFFull) << 48);
  g1.q[1] = ((td0 >> 16) & 0xFFFFull) | ((td1 & 0xFFFFFFFFull) << 16)
          | (((unsigned long long)(unsigned)tile_x & 0xFFFFull) << 48);
  g1.q[2] = ((unsigned long long)(unsigned)tile_y & 0xFFFFull)
          | ((st & 0xFFFFFFFFull) << 32);
  g1.q[3] = (st >> 32) & 0xFFFFull;
  i32x4 z4 = {0, 0, 0, 0};
#if defined(__clang_major__) && (__clang_major__ >= 23)
  i32x8 z8 = {0, 0, 0, 0, 0, 0, 0, 0};
  __builtin_amdgcn_tensor_load_to_lds(g0, g1.v, z4, z4, z8, 0);
#else
  __builtin_amdgcn_tensor_load_to_lds(g0, g1.v, z4, z4, 0);
#endif
}
#endif // KIM_HAS_TDM

// ---------------------------------------------------------------------------
// Generic batched GEMM: C = act(A @ B (+bias)), one wave per 16x16 C tile.
// A:[M,K] lda=K. B:[K,N] (transB=0) or [N,K] (transB=1). M,N masked,
// K zero-padded to 32 (via TDM tensor-dim OOB zero-fill).
// ---------------------------------------------------------------------------
__global__ void __launch_bounds__(32)
kim_gemm_wmma(const float* __restrict__ A, const float* __restrict__ B,
              const float* __restrict__ bias, float* __restrict__ C,
              int M, int N, int K, int lda, int ldb, int ldc,
              int transB, int act,
              long long sA, long long sB, long long sC)
{
  A += (long long)blockIdx.z * sA;
  B += (long long)blockIdx.z * sB;
  C += (long long)blockIdx.z * sC;
  const int lane = threadIdx.x;
  const int tm = blockIdx.y * 16, tn = blockIdx.x * 16;
  __shared__ unsigned short As[16 * 32];   // bf16 A tile, row-major [m][k]
  __shared__ unsigned short Bs[16 * 32];   // bf16 B tile, transposed [n][k]
  v8f acc = {0.f, 0.f, 0.f, 0.f, 0.f, 0.f, 0.f, 0.f};

#if KIM_HAS_TDM
  __shared__ float Af32[2][16 * 32];       // TDM staging, double buffered
  __shared__ float Bf32[2][16 * 32];
  auto issue_pair = [&](int kk, int buf) {
    // A tile: 32 contiguous k x 16 m-rows
    kim_tdm_load2d(A + (long long)tm * lda + kk, kim_lds_off(&Af32[buf][0]),
                   32, 16, (long long)K - kk, (long long)M - tm, lda);
    if (transB)   // B[N,K]: 32 contiguous k x 16 n-rows
      kim_tdm_load2d(B + (long long)tn * ldb + kk, kim_lds_off(&Bf32[buf][0]),
                     32, 16, (long long)K - kk, (long long)N - tn, ldb);
    else          // B[K,N]: 16 contiguous n x 32 k-rows
      kim_tdm_load2d(B + (long long)kk * ldb + tn, kim_lds_off(&Bf32[buf][0]),
                     16, 32, (long long)N - tn, (long long)K - kk, ldb);
  };
  const int nk = (K + 31) >> 5;
  issue_pair(0, 0);
  for (int ik = 0; ik < nk; ++ik) {
    const int buf = ik & 1;
    __builtin_amdgcn_s_wait_tensorcnt((short)0);   // current pair landed in LDS
    // f32 -> bf16 into WMMA fragment layout (no bounds checks: HW zero-fill)
#pragma unroll
    for (int t = 0; t < 16; ++t) {
      int e = t * 32 + lane;
      As[e] = kim_f2bf(Af32[buf][e]);
      if (transB) Bs[e] = kim_f2bf(Bf32[buf][e]);                       // [n][k]
      else        Bs[e] = kim_f2bf(Bf32[buf][(e & 31) * 16 + (e >> 5)]); // [k][n] -> [n][k]
    }
    __syncthreads();
    if (ik + 1 < nk) issue_pair((ik + 1) * 32, buf ^ 1);   // overlap DMA w/ MMA
#else
  for (int kk = 0; kk < K; kk += 32) {
    {
      int pm = tm + (lane & 15);
      if (kk + 32 < K && pm < M)
        __builtin_prefetch(&A[(long long)pm * lda + kk + 32], 0, 1);
    }
#pragma unroll
    for (int t = 0; t < 16; ++t) {
      int e = t * 32 + lane;
      int r = e >> 5, c = e & 31;
      int gm = tm + r, gk = kk + c;
      float v = (gm < M && gk < K) ? A[(long long)gm * lda + gk] : 0.f;
      As[e] = kim_f2bf(v);
    }
#pragma unroll
    for (int t = 0; t < 16; ++t) {
      int e = t * 32 + lane;
      int nn = e >> 5, c = e & 31;
      int gn = tn + nn, gk = kk + c;
      float v = 0.f;
      if (gn < N && gk < K)
        v = transB ? B[(long long)gn * ldb + gk] : B[(long long)gk * ldb + gn];
      Bs[e] = kim_f2bf(v);
    }
    __syncthreads();
#endif
    // fragments: element s of lane half h holds k = 16*h + s (contiguous), same order for A and B
    union { unsigned int u[8]; v16bf v; } fa, fb;
    const int mrow = lane & 15, half = lane >> 4;
    const int ka = half * 16;
#pragma unroll
    for (int j = 0; j < 8; ++j)
      fa.u[j] = *(const unsigned int*)&As[mrow * 32 + ka + 2 * j];
    const int kbb = half * 16;
#pragma unroll
    for (int j = 0; j < 8; ++j)
      fb.u[j] = *(const unsigned int*)&Bs[mrow * 32 + kbb + 2 * j];

    acc = __builtin_amdgcn_wmma_f32_16x16x32_bf16(false, fa.v, false, fb.v,
                                                  (short)0, acc, false, false);
    __syncthreads();
  }

  const int gn = tn + (lane & 15);
  if (gn < N) {
    const float bv = bias ? bias[gn] : 0.f;
#pragma unroll
    for (int r = 0; r < 8; ++r) {
      int gm = tm + r + ((lane >> 4) << 3);
      if (gm < M) {
        float v = acc[r] + bv;
        if (act == 1) v = tanhf(v);
        C[(long long)gm * ldc + gn] = v;
      }
    }
  }
}

// out[m] = sum_k A[m,k]*w[k] + b[0]
__global__ void kim_gemv(const float* __restrict__ A, const float* __restrict__ w,
                         const float* __restrict__ bias, float* __restrict__ out,
                         int M, int K)
{
  int m = blockIdx.x * blockDim.x + threadIdx.x;
  if (m >= M) return;
  __builtin_prefetch(w, 0, 3);
  const float* a = A + (long long)m * K;
  float s = bias[0];
  for (int k = 0; k < K; ++k) s += a[k] * w[k];
  out[m] = s;
}

// per-(sequence, head) self-attention, L=23, hd=20; lane = query position
__global__ void kim_mha23(const float* __restrict__ Q, const float* __restrict__ Kx,
                          const float* __restrict__ V, float* __restrict__ O,
                          float scale)
{
  long long base = (long long)blockIdx.x * WN * AD + blockIdx.y * HD;
  int q = threadIdx.x;
  if (q >= WN) return;
  float qv[HD], sc[WN];
  const float* qp = Q + base + (long long)q * AD;
#pragma unroll
  for (int d = 0; d < HD; ++d) qv[d] = qp[d];
  float mx = -1e30f;
  for (int k = 0; k < WN; ++k) {
    const float* kp = Kx + base + (long long)k * AD;
    float dd = 0.f;
#pragma unroll
    for (int d = 0; d < HD; ++d) dd += qv[d] * kp[d];
    dd *= scale;
    sc[k] = dd;
    mx = fmaxf(mx, dd);
  }
  float sum = 0.f;
  for (int k = 0; k < WN; ++k) { sc[k] = expf(sc[k] - mx); sum += sc[k]; }
  float inv = 1.f / sum;
  float* op = O + base + (long long)q * AD;
  for (int d = 0; d < HD; ++d) {
    float o = 0.f;
    for (int k = 0; k < WN; ++k) o += sc[k] * V[base + (long long)k * AD + d];
    op[d] = o * inv;
  }
}

// cc[b,f] = 0.001 * softmax_c(cross[b,f,:]) . nt_a0[b,:]
__global__ void kim_cross_candi(const float* __restrict__ cross,
                                const float* __restrict__ nta0,
                                float* __restrict__ cc)
{
  int idx = blockIdx.x * blockDim.x + threadIdx.x;
  if (idx >= BSZ * MC * WN) return;
  int b = idx / (MC * WN);
  const float* row = cross + (long long)idx * WN;
  const float* a0 = nta0 + b * WN;
  float mx = -1e30f;
  for (int c = 0; c < WN; ++c) mx = fmaxf(mx, row[c]);
  float s = 0.f, acc = 0.f;
  for (int c = 0; c < WN; ++c) {
    float e = expf(row[c] - mx);
    s += e; acc += e * a0[c];
  }
  cc[idx] = 0.001f * acc / s;
}

// out[r,:] = softmax_w(x[r,:] + y[r,:])
__global__ void kim_add_softmax(const float* __restrict__ x, const float* __restrict__ y,
                                float* __restrict__ out, int R, int W)
{
  int r = blockIdx.x * blockDim.x + threadIdx.x;
  if (r >= R) return;
  float mx = -1e30f;
  for (int w = 0; w < W; ++w) mx = fmaxf(mx, x[r * W + w] + y[r * W + w]);
  float s = 0.f;
  for (int w = 0; w < W; ++w) {
    float e = expf(x[r * W + w] + y[r * W + w] - mx);
    out[r * W + w] = e; s += e;
  }
  float inv = 1.f / s;
  for (int w = 0; w < W; ++w) out[r * W + w] *= inv;
}

// nt_att[b,n,c] = softmax_c( nt_a0[b,c] + sum_w cross[b,(n,w),c]*ct_att[b,n,w] )
__global__ void kim_ntatt(const float* __restrict__ cross, const float* __restrict__ ctatt,
                          const float* __restrict__ nta0, float* __restrict__ out)
{
  int r = blockIdx.x * blockDim.x + threadIdx.x;
  if (r >= BSZ * MC) return;
  int b = r / MC, n = r % MC;
  float v[WN];
  float mx = -1e30f;
  for (int c = 0; c < WN; ++c) {
    float s = nta0[b * WN + c];
    for (int w = 0; w < WN; ++w)
      s += cross[((long long)b * (MC * WN) + n * WN + w) * WN + c] * ctatt[r * WN + w];
    v[c] = s; mx = fmaxf(mx, s);
  }
  float sum = 0.f;
  for (int c = 0; c < WN; ++c) { v[c] = expf(v[c] - mx); sum += v[c]; }
  float inv = 1.f / sum;
  for (int c = 0; c < WN; ++c) out[r * WN + c] = v[c] * inv;
}

// nt_vecs[r=(b,n), d] = sum_c att[r,c]*nt[b,c,d]
__global__ void kim_ntvecs(const float* __restrict__ att, const float* __restrict__ nt,
                           float* __restrict__ out)
{
  int r = blockIdx.x;
  int b = r / MC;
  for (int d = threadIdx.x; d < AD; d += blockDim.x) {
    float s = 0.f;
    for (int c = 0; c < WN; ++c) s += att[r * WN + c] * nt[((long long)b * WN + c) * AD + d];
    out[(long long)r * AD + d] = s;
  }
}

// ct_vecs[r=(b,n), d] = sum_w att[r,w]*ct[(r,w),d]
__global__ void kim_ctvecs(const float* __restrict__ att, const float* __restrict__ ct,
                           float* __restrict__ out)
{
  int r = blockIdx.x;
  for (int d = threadIdx.x; d < AD; d += blockDim.x) {
    float s = 0.f;
    for (int w = 0; w < WN; ++w) s += att[r * WN + w] * ct[((long long)r * WN + w) * AD + d];
    out[(long long)r * AD + d] = s;
  }
}

// out[r,d] = sum_j softmax_j(s[r,j]) * X[b*sxb + n*sxn + e*sxe + j*sxj + d]
// where r decomposes as (b,n,e) over (d0,d1,d2), j over NJ neighbors.
__global__ void kim_softsum(const float* __restrict__ s, const float* __restrict__ X,
                            float* __restrict__ out,
                            int d0, int d1, int d2, int NJ, int D,
                            long long sxb, long long sxn, long long sxe, long long sxj)
{
  long long idx = (long long)blockIdx.x * blockDim.x + threadIdx.x;
  long long R = (long long)d0 * d1 * d2;
  if (idx >= R * D) return;
  long long r = idx / D;
  int d = (int)(idx % D);
  int e = (int)(r % d2);
  long long t = r / d2;
  int n = (int)(t % d1);
  int b = (int)(t / d1);
  const float* sp = s + r * NJ;
  float mx = -1e30f;
  for (int j = 0; j < NJ; ++j) mx = fmaxf(mx, sp[j]);
  float sum = 0.f, acc = 0.f;
  const float* xp = X + b * sxb + n * sxn + e * sxe + d;
  for (int j = 0; j < NJ; ++j) {
    float w = expf(sp[j] - mx);
    sum += w;
    acc += w * xp[j * sxj];
  }
  out[idx] = acc / sum;
}

// dst[r, 0:K1+K2] = concat(A1 row, A2 row); r = (i0,i1,i2,i3); strides are
// row-base element offsets (0 => broadcast along that index).
__global__ void kim_concat(const float* __restrict__ A1, const float* __restrict__ A2,
                           float* __restrict__ dst,
                           int d0, int d1, int d2, int d3, int K1, int K2,
                           long long a10, long long a11, long long a12, long long a13,
                           long long a20, long long a21, long long a22, long long a23)
{
  long long total = (long long)d0 * d1 * d2 * d3 * (K1 + K2);
  long long idx = (long long)blockIdx.x * blockDim.x + threadIdx.x;
  if (idx >= total) return;
  int KT = K1 + K2;
  long long r = idx / KT;
  int k = (int)(idx % KT);
  int i3 = (int)(r % d3); long long t = r / d3;
  int i2 = (int)(t % d2); t /= d2;
  int i1 = (int)(t % d1);
  int i0 = (int)(t / d1);
  float v;
  if (k < K1) v = A1[i0 * a10 + i1 * a11 + i2 * a12 + i3 * a13 + k];
  else        v = A2[i0 * a20 + i1 * a21 + i2 * a22 + i3 * a23 + (k - K1)];
  dst[idx] = v;
}

// full gcat cross for co_att: per (b,n,m1): scores over m2 entities -> softmax
// -> cross = sum softmax * Q. Q strides allow (b,n,m2) or (b,m2) broadcast.
__global__ void kim_gcat_cross(const float* __restrict__ ekw, const float* __restrict__ Qv,
                               float* __restrict__ cross,
                               long long sqb, long long sqn, long long sqm)
{
  int idx = blockIdx.x * blockDim.x + threadIdx.x;    // (b,n)*EN + m1
  if (idx >= BSZ * MC * EN) return;
  int r = idx / EN;
  int b = r / MC, n = r % MC;
  const float* ep = ekw + (long long)idx * ED;
  float sc[EN];
  float mx = -1e30f;
  for (int m2 = 0; m2 < EN; ++m2) {
    const float* qp = Qv + b * sqb + n * sqn + m2 * sqm;
    float s = 0.f;
    for (int d = 0; d < ED; ++d) s += ep[d] * qp[d];
    sc[m2] = s; mx = fmaxf(mx, s);
  }
  float sum = 0.f;
  for (int m2 = 0; m2 < EN; ++m2) { sc[m2] = expf(sc[m2] - mx); sum += sc[m2]; }
  float inv = 1.f / sum;
  float* cp = cross + (long long)idx * ED;
  for (int d = 0; d < ED; ++d) {
    float a = 0.f;
    for (int m2 = 0; m2 < EN; ++m2) a += sc[m2] * Qv[b * sqb + n * sqn + m2 * sqm + d];
    cp[d] = a * inv;
  }
}

static __device__ __forceinline__ float kim_block_reduce(float v, float* red) {
  int t = threadIdx.x;
  red[t] = v;
  __syncthreads();
  for (int s = blockDim.x / 2; s > 0; s >>= 1) {
    if (t < s) red[t] += red[t + s];
    __syncthreads();
  }
  float r = red[0];
  __syncthreads();
  return r;
}

// score[b,i] = dot( LN(sum_n news2[b,n,:]), LN(sum_n user2[b,n,:]) )
// (softmax over size-1 dim in _news_user_co makes na == ua == 1, faithfully).
__global__ void __launch_bounds__(256)
kim_final(const float* __restrict__ news2, const float* __restrict__ user2,
          const float* __restrict__ g1, const float* __restrict__ b1,
          const float* __restrict__ g2, const float* __restrict__ b2,
          float* __restrict__ out, int samp)
{
  int b = blockIdx.x;
  __shared__ float sn[AD];
  __shared__ float su[AD];
  __shared__ float red[256];
  for (int d = threadIdx.x; d < AD; d += blockDim.x) {
    float a = 0.f, c = 0.f;
    for (int n = 0; n < MC; ++n) {
      long long row = (long long)(b * MC + n) * AD + d;
      a += news2[row]; c += user2[row];
    }
    sn[d] = a; su[d] = c;
  }
  __syncthreads();
  float s1 = 0, s2 = 0, s3 = 0, s4 = 0;
  for (int d = threadIdx.x; d < AD; d += blockDim.x) {
    float a = sn[d], c = su[d];
    s1 += a; s2 += a * a; s3 += c; s4 += c * c;
  }
  s1 = kim_block_reduce(s1, red);
  s2 = kim_block_reduce(s2, red);
  s3 = kim_block_reduce(s3, red);
  s4 = kim_block_reduce(s4, red);
  float mn = s1 / AD, vn = s2 / AD - mn * mn;
  float mu = s3 / AD, vu = s4 / AD - mu * mu;
  float rn = rsqrtf(vn + 1e-5f), ru = rsqrtf(vu + 1e-5f);
  float dot = 0.f;
  for (int d = threadIdx.x; d < AD; d += blockDim.x) {
    float x = (sn[d] - mn) * rn * g1[d] + b1[d];
    float y = (su[d] - mu) * ru * g2[d] + b2[d];
    dot += x * y;
  }
  dot = kim_block_reduce(dot, red);
  if (threadIdx.x == 0) out[b * SNUM + samp] = dot;
}

// --------------------------- host-side helpers -----------------------------
static inline void launch_gemm(hipStream_t st, const float* A, const float* B,
                               const float* bias, float* C, int M, int N, int K,
                               int transB = 0, int act = 0, int batch = 1,
                               long long sA = 0, long long sB = 0, long long sC = 0)
{
  int lda = K;
  int ldb = transB ? K : N;
  int ldc = N;
  dim3 g((unsigned)((N + 15) / 16), (unsigned)((M + 15) / 16), (unsigned)batch);
  kim_gemm_wmma<<<g, 32, 0, st>>>(A, B, bias, C, M, N, K, lda, ldb, ldc,
                                  transB, act, sA, sB, sC);
}

static inline void launch_gemv(hipStream_t st, const float* A, const float* w,
                               const float* b, float* o, int M, int K)
{
  kim_gemv<<<(M + 255) / 256, 256, 0, st>>>(A, w, b, o, M, K);
}

static inline void launch_softsum(hipStream_t st, const float* s, const float* X, float* out,
                                  int d0, int d1, int d2, int NJ, int D,
                                  long long sxb, long long sxn, long long sxe, long long sxj)
{
  long long total = (long long)d0 * d1 * d2 * D;
  kim_softsum<<<(unsigned)((total + 255) / 256), 256, 0, st>>>(s, X, out, d0, d1, d2, NJ, D,
                                                              sxb, sxn, sxe, sxj);
}

static inline void launch_concat(hipStream_t st, const float* A1, const float* A2, float* dst,
                                 int d0, int d1, int d2, int d3, int K1, int K2,
                                 long long a10, long long a11, long long a12, long long a13,
                                 long long a20, long long a21, long long a22, long long a23)
{
  long long total = (long long)d0 * d1 * d2 * d3 * (K1 + K2);
  kim_concat<<<(unsigned)((total + 255) / 256), 256, 0, st>>>(A1, A2, dst, d0, d1, d2, d3,
                                                              K1, K2, a10, a11, a12, a13,
                                                              a20, a21, a22, a23);
}

extern "C" void kernel_launch(void* const* d_in, const int* in_sizes, int n_in,
                              void* d_out, int out_size, void* d_ws, size_t ws_size,
                              hipStream_t stream)
{
  (void)in_sizes; (void)n_in; (void)out_size; (void)ws_size;
  auto in = [&](int i) { return (const float*)d_in[i]; };

  // params flattened in jax pytree (sorted-key) order
  const float* co_a1b  = in(0);
  const float* co_a1w  = in(1);
  const float* co_a2b  = in(2);
  const float* co_a2w  = in(3);
  const float* co_kb   = in(4);
  const float* co_kw   = in(5);
  const float* gat_b1  = in(6);
  const float* gat_b2  = in(7);
  const float* gat_w1  = in(8);
  const float* gat_w2  = in(9);
  const float* gatfc_b = in(10);
  const float* gatfc_w = in(11);
  const float* gc_a1b  = in(12);
  const float* gc_a1w  = in(13);
  const float* gc_a2b  = in(14);
  const float* gc_a2w  = in(15);
  const float* gcfc_b  = in(18);
  const float* gcfc_w  = in(19);
  const float* merge_b = in(20);
  const float* merge_w = in(21);
  const float* n1b     = in(24);
  const float* n1g     = in(25);
  const float* n2b     = in(26);
  const float* n2g     = in(27);
  const float* ca1b    = in(36);
  const float* ca1w    = in(37);
  const float* ca2b    = in(38);
  const float* ca2w    = in(39);
  const float* k2w     = in(40);
  const float* k3w     = in(41);
  const float* na1b    = in(42);
  const float* na1w    = in(43);
  const float* na2b    = in(44);
  const float* na2w    = in(45);
  const float* q2w     = in(46);
  const float* q3w     = in(47);
  const float* v2w     = in(48);
  const float* v3w     = in(49);
  const float* cand_word      = in(50);
  const float* clicked_word   = in(51);
  const float* cand_entity    = in(52);
  const float* clicked_entity = in(53);
  const float* cand_neigh     = in(54);
  const float* clicked_neigh  = in(55);
  float* out = (float*)d_out;

  const float scale = 0.2236067977f;           // 1/sqrt(HD)
  const int TOK_C = BSZ * MC * WN;             // 18400 clicked tokens
  const int TOK_N = BSZ * WN;                  // 368 candidate tokens

  // ---- workspace layout (floats) ----
  float* W = (float*)d_ws;
  size_t off = 0;
  auto alloc = [&](size_t n) { float* p = W + off; off += n; return p; };
  float* ct    = alloc((size_t)TOK_C * AD);    // [B,50,23,400]
  float* ctav  = alloc((size_t)TOK_C * 200);   // [B,50,23,200]
  float* ctat0 = alloc((size_t)TOK_C);         // [B,50,23]
  float* ucan0 = alloc((size_t)4000 * ED);     // [B,50,5,100]
  const size_t sbase = off;                    // reusable arena

  // ============ clicked-side semantic (loop-invariant, hoisted) ============
  float* Qc = W + sbase;
  float* Kc = Qc + (size_t)TOK_C * AD;
  float* Vc = Kc + (size_t)TOK_C * AD;
  launch_gemm(stream, clicked_word, q3w, nullptr, Qc, TOK_C, AD, WD);
  launch_gemm(stream, clicked_word, k3w, nullptr, Kc, TOK_C, AD, WD);
  launch_gemm(stream, clicked_word, v3w, nullptr, Vc, TOK_C, AD, WD);
  kim_mha23<<<dim3(BSZ * MC, HEADS), 32, 0, stream>>>(Qc, Kc, Vc, ct, scale);
  launch_gemm(stream, ct, ca1w, ca1b, ctav, TOK_C, 200, AD, 0, /*tanh*/1);
  launch_gemv(stream, ctav, ca2w, ca2b, ctat0, TOK_C, 200);

  // ============ clicked-side knowledge GAT (loop-invariant) ============
  off = sbase;
  {
    float* gh   = alloc((size_t)20000 * ED);
    float* gs   = alloc(20000);
    float* ugat = alloc((size_t)4000 * ED);
    float* ucat = alloc((size_t)4000 * 2 * ED);
    launch_gemm(stream, clicked_neigh, gat_w1, gat_b1, gh, 20000, ED, ED, 0, 1);
    launch_gemv(stream, gh, gat_w2, gat_b2, gs, 20000, ED);
    launch_softsum(stream, gs, clicked_neigh, ugat, BSZ, MC, EN, NN, ED,
                   125000, 2500, 500, 100);
    launch_concat(stream, ugat, clicked_entity, ucat, BSZ, MC, EN, 1, ED, ED,
                  25000, 500, 100, 0, 25000, 500, 100, 0);
    launch_gemm(stream, ucat, gatfc_w, gatfc_b, ucan0, 4000, ED, 2 * ED);
  }

  // ============ per-candidate-sample loop ============
  for (int i = 0; i < SNUM; ++i) {
    off = sbase;
    float* ntq   = alloc((size_t)TOK_N * AD);
    float* ntk   = alloc((size_t)TOK_N * AD);
    float* ntv   = alloc((size_t)TOK_N * AD);
    float* nt    = alloc((size_t)TOK_N * AD);
    float* ntav  = alloc((size_t)TOK_N * 200);
    float* nta0  = alloc(TOK_N);
    float* crs   = alloc((size_t)BSZ * MC * WN * WN);   // [B,1150,23]
    float* cc    = alloc(TOK_C);
    float* ctatt = alloc(TOK_C);
    float* ntatt = alloc(TOK_C);
    float* ntvec = alloc((size_t)BSZ * MC * AD);
    float* ctvec = alloc((size_t)BSZ * MC * AD);
    float* ngh   = alloc((size_t)400 * ED);
    float* ngs   = alloc(400);
    float* ngat  = alloc((size_t)80 * ED);
    float* ncat  = alloc((size_t)80 * 2 * ED);
    float* ncan0 = alloc((size_t)80 * ED);
    float* hcat  = alloc((size_t)20000 * 2 * ED);
    float* hbuf  = alloc((size_t)20000 * ED);
    float* hs    = alloc(20000);
    float* nsoh  = alloc((size_t)4000 * ED);
    float* usoh  = alloc((size_t)4000 * ED);
    float* nvcat = alloc((size_t)4000 * 2 * ED);
    float* newv  = alloc((size_t)4000 * ED);
    float* uvcat = alloc((size_t)4000 * 2 * ED);
    float* userv = alloc((size_t)4000 * ED);
    float* ekwn  = alloc((size_t)4000 * ED);
    float* crsn  = alloc((size_t)4000 * ED);
    float* ekwu  = alloc((size_t)4000 * ED);
    float* crsu  = alloc((size_t)4000 * ED);
    float* h2cat = alloc((size_t)4000 * 2 * ED);
    float* h2    = alloc((size_t)4000 * ED);
    float* s2n   = alloc(4000);
    float* s2u   = alloc(4000);
    float* nev   = alloc((size_t)BSZ * MC * ED);
    float* uev   = alloc((size_t)BSZ * MC * ED);
    float* mcn   = alloc((size_t)BSZ * MC * (AD + ED));
    float* mcu   = alloc((size_t)BSZ * MC * (AD + ED));
    float* news2 = alloc((size_t)BSZ * MC * AD);
    float* user2 = alloc((size_t)BSZ * MC * AD);

    // ---- candidate semantic ----
    const float* Xn = cand_word + (long long)i * WN * WD;     // strided slice [:,i]
    const long long sXn = (long long)SNUM * WN * WD;
    launch_gemm(stream, Xn, q2w, nullptr, ntq, WN, AD, WD, 0, 0, BSZ, sXn, 0, (long long)WN * AD);
    launch_gemm(stream, Xn, k2w, nullptr, ntk, WN, AD, WD, 0, 0, BSZ, sXn, 0, (long long)WN * AD);
    launch_gemm(stream, Xn, v2w, nullptr, ntv, WN, AD, WD, 0, 0, BSZ, sXn, 0, (long long)WN * AD);
    kim_mha23<<<dim3(BSZ, HEADS), 32, 0, stream>>>(ntq, ntk, ntv, nt, scale);
    launch_gemm(stream, nt, na1w, na1b, ntav, TOK_N, 200, AD, 0, 1);
    launch_gemv(stream, ntav, na2w, na2b, nta0, TOK_N, 200);
    // cross[b] = ct_av[b] @ nt_av[b]^T  (batched, transB)
    launch_gemm(stream, ctav, ntav, nullptr, crs, MC * WN, WN, 200, /*transB*/1, 0,
                BSZ, (long long)MC * WN * 200, (long long)WN * 200, (long long)MC * WN * WN);
    kim_cross_candi<<<(BSZ * MC * WN + 255) / 256, 256, 0, stream>>>(crs, nta0, cc);
    kim_add_softmax<<<(BSZ * MC + 255) / 256, 256, 0, stream>>>(ctat0, cc, ctatt, BSZ * MC, WN);
    kim_ntatt<<<(BSZ * MC + 255) / 256, 256, 0, stream>>>(crs, ctatt, nta0, ntatt);
    kim_ntvecs<<<BSZ * MC, 128, 0, stream>>>(ntatt, nt, ntvec);
    kim_ctvecs<<<BSZ * MC, 128, 0, stream>>>(ctatt, ct, ctvec);

    // ---- candidate knowledge GAT ----
    const float* NOi = cand_neigh + (long long)i * EN * NN * ED;    // [B,(i),5,5,100]
    const float* NEi = cand_entity + (long long)i * EN * ED;        // [B,(i),5,100]
    launch_gemm(stream, NOi, gat_w1, gat_b1, ngh, EN * NN, ED, ED, 0, 1,
                BSZ, (long long)SNUM * EN * NN * ED, 0, (long long)EN * NN * ED);
    launch_gemv(stream, ngh, gat_w2, gat_b2, ngs, BSZ * EN * NN, ED);
    launch_softsum(stream, ngs, NOi, ngat, BSZ, EN, 1, NN, ED, 12500, 500, 0, 100);
    launch_concat(stream, ngat, NEi, ncat, BSZ, EN, 1, 1, ED, ED,
                  500, 100, 0, 0, 2500, 100, 0, 0);
    launch_gemm(stream, ncat, gatfc_w, gatfc_b, ncan0, BSZ * EN, ED, 2 * ED);

    // ---- news_oh gcat (cross == user_can0 by broadcast identity) ----
    launch_concat(stream, NOi, ucan0, hcat, BSZ, MC, EN, NN, ED, ED,
                  12500, 0, 500, 100, 25000, 500, 100, 0);
    launch_gemm(stream, hcat, gc_a1w, gc_a1b, hbuf, 20000, ED, 2 * ED, 0, 1);
    launch_gemv(stream, hbuf, gc_a2w, gc_a2b, hs, 20000, ED);
    launch_softsum(stream, hs, NOi, nsoh, BSZ, MC, EN, NN, ED, 12500, 0, 500, 100);

    // ---- user_oh gcat (cross == new_can0 by broadcast identity) ----
    launch_concat(stream, clicked_neigh, ncan0, hcat, BSZ, MC, EN, NN, ED, ED,
                  125000, 2500, 500, 100, 500, 0, 100, 0);
    launch_gemm(stream, hcat, gc_a1w, gc_a1b, hbuf, 20000, ED, 2 * ED, 0, 1);
    launch_gemv(stream, hbuf, gc_a2w, gc_a2b, hs, 20000, ED);
    launch_softsum(stream, hs, clicked_neigh, usoh, BSZ, MC, EN, NN, ED,
                   125000, 2500, 500, 100);

    // ---- fuse onehop into entity vectors ----
    launch_concat(stream, NEi, nsoh, nvcat, BSZ, MC, EN, 1, ED, ED,
                  2500, 0, 100, 0, 25000, 500, 100, 0);
    launch_gemm(stream, nvcat, gcfc_w, gcfc_b, newv, 4000, ED, 2 * ED);
    launch_concat(stream, clicked_entity, usoh, uvcat, BSZ, MC, EN, 1, ED, ED,
                  25000, 500, 100, 0, 25000, 500, 100, 0);
    launch_gemm(stream, uvcat, gcfc_w, gcfc_b, userv, 4000, ED, 2 * ED);

    // ---- co_att gcat on news side (query = clicked_entity) ----
    launch_gemm(stream, newv, co_kw, co_kb, ekwn, 4000, ED, ED);
    kim_gcat_cross<<<(4000 + 255) / 256, 256, 0, stream>>>(ekwn, clicked_entity, crsn,
                                                           25000, 500, 100);
    launch_concat(stream, newv, crsn, h2cat, BSZ, MC, EN, 1, ED, ED,
                  25000, 500, 100, 0, 25000, 500, 100, 0);
    launch_gemm(stream, h2cat, co_a1w, co_a1b, h2, 4000, ED, 2 * ED, 0, 1);
    launch_gemv(stream, h2, co_a2w, co_a2b, s2n, 4000, ED);
    launch_softsum(stream, s2n, newv, nev, BSZ, MC, 1, EN, ED, 25000, 500, 0, 100);

    // ---- co_att gcat on user side (query = new_entity, bcast over clicks) ----
    launch_gemm(stream, userv, co_kw, co_kb, ekwu, 4000, ED, ED);
    kim_gcat_cross<<<(4000 + 255) / 256, 256, 0, stream>>>(ekwu, NEi, crsu,
                                                           2500, 0, 100);
    launch_concat(stream, userv, crsu, h2cat, BSZ, MC, EN, 1, ED, ED,
                  25000, 500, 100, 0, 25000, 500, 100, 0);
    launch_gemm(stream, h2cat, co_a1w, co_a1b, h2, 4000, ED, 2 * ED, 0, 1);
    launch_gemv(stream, h2, co_a2w, co_a2b, s2u, 4000, ED);
    launch_softsum(stream, s2u, userv, uev, BSZ, MC, 1, EN, ED, 25000, 500, 0, 100);

    // ---- merge + final score ----
    launch_concat(stream, ntvec, nev, mcn, BSZ, MC, 1, 1, AD, ED,
                  20000, 400, 0, 0, 5000, 100, 0, 0);
    launch_gemm(stream, mcn, merge_w, merge_b, news2, BSZ * MC, AD, AD + ED);
    launch_concat(stream, ctvec, uev, mcu, BSZ, MC, 1, 1, AD, ED,
                  20000, 400, 0, 0, 5000, 100, 0, 0);
    launch_gemm(stream, mcu, merge_w, merge_b, user2, BSZ * MC, AD, AD + ED);
    kim_final<<<BSZ, 256, 0, stream>>>(news2, user2, n1g, n1b, n2g, n2b, out, i);
  }
}
